// GridSelfAttention_63668595195922
// MI455X (gfx1250) — matrix operand order repaired and checked by reading, not compiled, not run on hardware
//
#include <hip/hip_runtime.h>
#include <stdint.h>

namespace {

constexpr int kN  = 384;
constexpr int kC  = 128;
constexpr int kH  = 4;
constexpr int kD  = 32;
constexpr int kNN = kN * kN;          // 147456
constexpr int kWTStride = 136;        // padded LDS row stride (bf16 elems) for W^T
constexpr int kPStride  = 392;        // padded LDS row stride (bf16 elems) for P

typedef __attribute__((ext_vector_type(16))) __bf16 v16bf;
typedef __attribute__((ext_vector_type(8)))  __bf16 v8bf;
typedef __attribute__((ext_vector_type(4)))  __bf16 v4bf;
typedef __attribute__((ext_vector_type(8)))  float  v8f;

// Load a 16-element bf16 fragment from two 16B-aligned 8-element chunks.
__device__ __forceinline__ v16bf ld16(const __bf16* p0, const __bf16* p1) {
  v8bf a = *(const v8bf*)p0;
  v8bf b = *(const v8bf*)p1;
  v16bf r;
#pragma unroll
  for (int i = 0; i < 8; ++i) { r[i] = a[i]; r[i + 8] = b[i]; }
  return r;
}

__device__ __forceinline__ v8f wmma_bf16(v16bf a, v16bf b, v8f c) {
  // (neg_a, A, neg_b, B, c_mod, C, reuse_a, reuse_b)
  return __builtin_amdgcn_wmma_f32_16x16x32_bf16(false, a, false, b, (short)0, c,
                                                 false, false);
}

// ---------------------------------------------------------------------------
// K1: LayerNorm over C + per-position head bias.  One wave32 per (q,k) row.
// ---------------------------------------------------------------------------
__global__ __launch_bounds__(256) void k_ln_bias(
    const float* __restrict__ pair, const float* __restrict__ lnw,
    const float* __restrict__ lnb, const float* __restrict__ wbias,
    __bf16* __restrict__ xout, float* __restrict__ biasout) {
  const int row  = blockIdx.x * 8 + (threadIdx.x >> 5);
  const int lane = threadIdx.x & 31;

  const float4 xv = *(const float4*)(pair + (size_t)row * kC + lane * 4);
  float s = xv.x + xv.y + xv.z + xv.w;
#pragma unroll
  for (int m = 1; m < 32; m <<= 1) s += __shfl_xor(s, m, 32);
  const float mu = s * (1.0f / kC);

  float dx[4] = {xv.x - mu, xv.y - mu, xv.z - mu, xv.w - mu};
  float vs = dx[0] * dx[0] + dx[1] * dx[1] + dx[2] * dx[2] + dx[3] * dx[3];
#pragma unroll
  for (int m = 1; m < 32; m <<= 1) vs += __shfl_xor(vs, m, 32);
  const float rstd = rsqrtf(vs * (1.0f / kC) + 1e-5f);

  const float4 wv = *(const float4*)(lnw + lane * 4);
  const float4 bv = *(const float4*)(lnb + lane * 4);
  float xn[4];
  xn[0] = dx[0] * rstd * wv.x + bv.x;
  xn[1] = dx[1] * rstd * wv.y + bv.y;
  xn[2] = dx[2] * rstd * wv.z + bv.z;
  xn[3] = dx[3] * rstd * wv.w + bv.w;

  v4bf xo;
#pragma unroll
  for (int j = 0; j < 4; ++j) xo[j] = (__bf16)xn[j];
  *(v4bf*)(xout + (size_t)row * kC + lane * 4) = xo;

  // bias[h, q, k] = sum_c xn[c] * wbias[c, h]
  float bb[4] = {0.f, 0.f, 0.f, 0.f};
#pragma unroll
  for (int j = 0; j < 4; ++j) {
    const float4 wb = *(const float4*)(wbias + (size_t)(lane * 4 + j) * kH);
    bb[0] += xn[j] * wb.x;
    bb[1] += xn[j] * wb.y;
    bb[2] += xn[j] * wb.z;
    bb[3] += xn[j] * wb.w;
  }
#pragma unroll
  for (int jj = 0; jj < 4; ++jj) {
#pragma unroll
    for (int m = 1; m < 32; m <<= 1) bb[jj] += __shfl_xor(bb[jj], m, 32);
  }
  if (lane == 0) {
#pragma unroll
    for (int jj = 0; jj < 4; ++jj) biasout[(size_t)jj * kNN + row] = bb[jj];
  }
}

// ---------------------------------------------------------------------------
// K2: fused projections  X[NN,128] @ {wq,wk,wv,wg}[128,128]  (blockIdx.y picks
// the weight).  Block: 64 rows x 128 cols, 8 waves, 4 WMMA tiles each.
// W is transposed into LDS so B-fragments are contiguous per lane.
// ---------------------------------------------------------------------------
__global__ __launch_bounds__(256) void k_proj(
    const __bf16* __restrict__ x, const float* __restrict__ wq,
    const float* __restrict__ wk, const float* __restrict__ wv,
    const float* __restrict__ wg, __bf16* __restrict__ qb,
    __bf16* __restrict__ kb, __bf16* __restrict__ vt, __bf16* __restrict__ gb) {
  __shared__ __bf16 wt[kC * kWTStride];
  const int wsel = blockIdx.y;
  const float* W = (wsel == 0) ? wq : (wsel == 1) ? wk : (wsel == 2) ? wv : wg;
  for (int e = threadIdx.x; e < kC * kC; e += 256) {
    const int kk = e >> 7, col = e & 127;            // W[k][col] -> WT[col][k]
    wt[col * kWTStride + kk] = (__bf16)W[e];
  }
  __syncthreads();

  const int wave = threadIdx.x >> 5, lane = threadIdx.x & 31;
  const int half = lane >> 4, l15 = lane & 15;
  const int mt = wave & 3, ng = (wave >> 2) * 4;
  const int rowBase = blockIdx.x * 64 + mt * 16;
  const __bf16* arow = x + (size_t)(rowBase + l15) * kC + half * 16;

  v8f acc[4];
#pragma unroll
  for (int t = 0; t < 4; ++t)
#pragma unroll
    for (int i = 0; i < 8; ++i) acc[t][i] = 0.0f;

#pragma unroll
  for (int ks = 0; ks < 4; ++ks) {
    const __bf16* pa = arow + ks * 32;
    v16bf a = ld16(pa, pa + 8);
#pragma unroll
    for (int t = 0; t < 4; ++t) {
      const __bf16* pb =
          &wt[((ng + t) * 16 + l15) * kWTStride + ks * 32 + half * 16];
      acc[t] = wmma_bf16(a, ld16(pb, pb + 8), acc[t]);
    }
  }

#pragma unroll
  for (int t = 0; t < 4; ++t) {
#pragma unroll
    for (int i = 0; i < 8; ++i) {
      const int r = rowBase + half * 8 + i;          // row in [NN)
      const int c = (ng + t) * 16 + l15;             // col in [128)
      const float v = acc[t][i];
      const int bi = r / kN, ni = r % kN;
      const int hh = c >> 5, dd = c & 31;
      if (wsel == 0) {        // q, pre-scaled by D^-0.5, layout [b,h,n,d]
        qb[(((size_t)(bi * kH + hh)) * kN + ni) * kD + dd] =
            (__bf16)(v * 0.17677669529663687f);
      } else if (wsel == 1) { // k, layout [b,h,n,d]
        kb[(((size_t)(bi * kH + hh)) * kN + ni) * kD + dd] = (__bf16)v;
      } else if (wsel == 2) { // v, TRANSPOSED layout [b,h,d,n]
        vt[(((size_t)(bi * kH + hh)) * kD + dd) * kN + ni] = (__bf16)v;
      } else {                // gate = sigmoid, layout [NN,128]
        gb[(size_t)r * kC + c] = (__bf16)(1.0f / (1.0f + __expf(-v)));
      }
    }
  }
}

// ---------------------------------------------------------------------------
// K3: attention for one (b, h, 64 q-rows) per block; 4 waves, 16 q-rows each.
// QK^T: 24 WMMAs (D=32 == one K step), bias folded in as the C operand.
// Softmax in registers (16-lane shfl reductions), exp'd P staged in LDS bf16,
// PV: 12 K-steps x 2 d-tiles.  Epilogue: /rowsum, * gate -> og (bf16).
// ---------------------------------------------------------------------------
__global__ __launch_bounds__(128) void k_attn(
    const __bf16* __restrict__ qb, const __bf16* __restrict__ kb,
    const __bf16* __restrict__ vt, const __bf16* __restrict__ gb,
    const float* __restrict__ biasb, const unsigned char* __restrict__ mask,
    __bf16* __restrict__ og) {
  extern __shared__ __bf16 pl[];
  const int wave = threadIdx.x >> 5, lane = threadIdx.x & 31;
  const int half = lane >> 4, l15 = lane & 15;
  const int b = blockIdx.z, h = blockIdx.y;
  const int qBase = blockIdx.x * 64 + wave * 16;
  const size_t bh = (size_t)(b * kH + h);
  const __bf16* qp = qb + bh * kN * kD;   // [n,d]
  const __bf16* kp = kb + bh * kN * kD;   // [n,d]
  const __bf16* vp = vt + bh * kD * kN;   // [d,n]
  const float* bias_h = biasb + (size_t)h * kNN;

  const __bf16* pa0 = qp + (size_t)(qBase + l15) * kD + half * 16;
  const v16bf aq = ld16(pa0, pa0 + 8);

  v8f s[24];
#pragma unroll
  for (int t = 0; t < 24; ++t) {
    const int c = t * 16 + l15;                      // key column
    v8f cb;                                          // bias as WMMA C operand
#pragma unroll
    for (int i = 0; i < 8; ++i)
      cb[i] = bias_h[(size_t)(qBase + half * 8 + i) * kN + c];
    const __bf16* pk = kp + (size_t)c * kD + half * 16;
    s[t] = wmma_bf16(aq, ld16(pk, pk + 8), cb);
    if (mask[(size_t)b * kN + c] == 0) {
#pragma unroll
      for (int i = 0; i < 8; ++i) s[t][i] = -1e9f;
    }
  }

  // row max (rows i / 8+i live in lane halves; 16-lane xor reduce)
  float m[8];
#pragma unroll
  for (int i = 0; i < 8; ++i) m[i] = s[0][i];
#pragma unroll
  for (int t = 1; t < 24; ++t)
#pragma unroll
    for (int i = 0; i < 8; ++i) m[i] = fmaxf(m[i], s[t][i]);
#pragma unroll
  for (int i = 0; i < 8; ++i) {
    m[i] = fmaxf(m[i], __shfl_xor(m[i], 1, 32));
    m[i] = fmaxf(m[i], __shfl_xor(m[i], 2, 32));
    m[i] = fmaxf(m[i], __shfl_xor(m[i], 4, 32));
    m[i] = fmaxf(m[i], __shfl_xor(m[i], 8, 32));
  }

  float rs[8] = {0.f, 0.f, 0.f, 0.f, 0.f, 0.f, 0.f, 0.f};
  __bf16* slab = pl + (size_t)wave * 16 * kPStride;
#pragma unroll
  for (int t = 0; t < 24; ++t) {
#pragma unroll
    for (int i = 0; i < 8; ++i) {
      const float e = __expf(s[t][i] - m[i]);
      rs[i] += e;
      slab[(half * 8 + i) * kPStride + t * 16 + l15] = (__bf16)e;
    }
  }
#pragma unroll
  for (int i = 0; i < 8; ++i) {
    rs[i] += __shfl_xor(rs[i], 1, 32);
    rs[i] += __shfl_xor(rs[i], 2, 32);
    rs[i] += __shfl_xor(rs[i], 4, 32);
    rs[i] += __shfl_xor(rs[i], 8, 32);
  }
  __syncthreads();  // order LDS P writes before fragment reads

  v8f o0, o1;
#pragma unroll
  for (int i = 0; i < 8; ++i) { o0[i] = 0.0f; o1[i] = 0.0f; }
#pragma unroll
  for (int ks = 0; ks < 12; ++ks) {
    const __bf16* pp = slab + l15 * kPStride + ks * 32 + half * 16;
    const v16bf ap = ld16(pp, pp + 8);
    const __bf16* pv0 = vp + (size_t)l15 * kN + ks * 32 + half * 16;
    o0 = wmma_bf16(ap, ld16(pv0, pv0 + 8), o0);
    const __bf16* pv1 = vp + (size_t)(16 + l15) * kN + ks * 32 + half * 16;
    o1 = wmma_bf16(ap, ld16(pv1, pv1 + 8), o1);
  }

#pragma unroll
  for (int i = 0; i < 8; ++i) {
    const int qr = qBase + half * 8 + i;
    const float inv = 1.0f / rs[i];
    {
      const size_t gi = ((size_t)b * kN + qr) * kC + h * kD + l15;
      og[gi] = (__bf16)(o0[i] * inv * (float)gb[gi]);
    }
    {
      const size_t gi = ((size_t)b * kN + qr) * kC + h * kD + 16 + l15;
      og[gi] = (__bf16)(o1[i] * inv * (float)gb[gi]);
    }
  }
}

// ---------------------------------------------------------------------------
// K4: out = (o * gate) @ wo  ->  fp32.  Same tiling as K2.
// ---------------------------------------------------------------------------
__global__ __launch_bounds__(256) void k_outproj(const __bf16* __restrict__ og,
                                                 const float* __restrict__ wo,
                                                 float* __restrict__ out) {
  __shared__ __bf16 wt[kC * kWTStride];
  for (int e = threadIdx.x; e < kC * kC; e += 256) {
    const int kk = e >> 7, col = e & 127;
    wt[col * kWTStride + kk] = (__bf16)wo[e];
  }
  __syncthreads();

  const int wave = threadIdx.x >> 5, lane = threadIdx.x & 31;
  const int half = lane >> 4, l15 = lane & 15;
  const int mt = wave & 3, ng = (wave >> 2) * 4;
  const int rowBase = blockIdx.x * 64 + mt * 16;
  const __bf16* arow = og + (size_t)(rowBase + l15) * kC + half * 16;

  v8f acc[4];
#pragma unroll
  for (int t = 0; t < 4; ++t)
#pragma unroll
    for (int i = 0; i < 8; ++i) acc[t][i] = 0.0f;

#pragma unroll
  for (int ks = 0; ks < 4; ++ks) {
    const __bf16* pa = arow + ks * 32;
    v16bf a = ld16(pa, pa + 8);
#pragma unroll
    for (int t = 0; t < 4; ++t) {
      const __bf16* pb =
          &wt[((ng + t) * 16 + l15) * kWTStride + ks * 32 + half * 16];
      acc[t] = wmma_bf16(a, ld16(pb, pb + 8), acc[t]);
    }
  }

#pragma unroll
  for (int t = 0; t < 4; ++t) {
#pragma unroll
    for (int i = 0; i < 8; ++i) {
      const int r = rowBase + half * 8 + i;
      const int c = (ng + t) * 16 + l15;
      out[(size_t)r * kC + c] = acc[t][i];
    }
  }
}

}  // namespace

extern "C" void kernel_launch(void* const* d_in, const int* in_sizes, int n_in,
                              void* d_out, int out_size, void* d_ws,
                              size_t ws_size, hipStream_t stream) {
  (void)in_sizes; (void)n_in; (void)out_size; (void)ws_size;
  const float* pair          = (const float*)d_in[0];
  const unsigned char* mask  = (const unsigned char*)d_in[1];  // bool
  const float* lnw           = (const float*)d_in[2];
  const float* lnb           = (const float*)d_in[3];
  const float* wbias         = (const float*)d_in[4];
  const float* wq            = (const float*)d_in[5];
  const float* wk            = (const float*)d_in[6];
  const float* wv            = (const float*)d_in[7];
  const float* wg            = (const float*)d_in[8];
  const float* wo            = (const float*)d_in[9];
  float* out                 = (float*)d_out;

  const size_t szBf = (size_t)kNN * kC * sizeof(__bf16);  // 36 MiB per buffer
  char* ws    = (char*)d_ws;
  __bf16* xb  = (__bf16*)ws;                 // LN output; reused for o*gate
  __bf16* qb  = (__bf16*)(ws + szBf);
  __bf16* kb  = (__bf16*)(ws + 2 * szBf);
  __bf16* vt  = (__bf16*)(ws + 3 * szBf);
  __bf16* gb  = (__bf16*)(ws + 4 * szBf);
  float* bias = (float*)(ws + 5 * szBf);     // [H, N, N] fp32

  k_ln_bias<<<dim3(kNN / 8), dim3(256), 0, stream>>>(pair, lnw, lnb, wbias, xb,
                                                     bias);
  k_proj<<<dim3(kNN / 64, 4), dim3(256), 0, stream>>>(xb, wq, wk, wv, wg, qb,
                                                      kb, vt, gb);
  k_attn<<<dim3(kN / 64, kH, kN), dim3(128),
           4 * 16 * kPStride * sizeof(__bf16), stream>>>(qb, kb, vt, gb, bias,
                                                         mask, xb);
  k_outproj<<<dim3(kNN / 64), dim3(256), 0, stream>>>(xb, wo, out);
}
